// RGCN_72224170049620
// MI455X (gfx1250) — compile-verified, matrix operand order checked by reading, not run on hardware
//
#include <hip/hip_runtime.h>
#include <hip/hip_bf16.h>
#include <math.h>

// Problem constants from the reference
#define NN 2048
#define TT 200
#define DD 16
#define HH 64
#define KSPLIT 4

typedef __attribute__((ext_vector_type(16))) __bf16 v16bf;
typedef __attribute__((ext_vector_type(8)))  __bf16 v8bf;
typedef __attribute__((ext_vector_type(8)))  float  v8f;

union bfrag { v16bf v; v8bf h[2]; };

__device__ __forceinline__ float sigmoidf(float x) { return 1.0f / (1.0f + __expf(-x)); }

// ---------------------------------------------------------------------------
// One-time: convert A (fp32, 16MB) -> bf16 (8MB, L2-resident for all 200 steps)
// ---------------------------------------------------------------------------
__global__ void convertA_kernel(const float* __restrict__ A, __bf16* __restrict__ Abf, int n) {
    int gid = blockIdx.x * blockDim.x + threadIdx.x;
    if (gid < n) Abf[gid] = (__bf16)A[gid];
}

// ---------------------------------------------------------------------------
// Per step: G = state @ W_graph + b, written TRANSPOSED as bf16 Gt[128][2048].
// cols 0..63 = h-projection, 64..127 = c-projection (fused so the big GEMM
// reads A once for both). m is fastest in gid -> coalesced Gt writes, and n is
// uniform per 256-thread block (no divergence on the h/c select).
// ---------------------------------------------------------------------------
__global__ void graph_pre_kernel(const float* __restrict__ h_cur, const float* __restrict__ c_cur,
                                 const float* __restrict__ Wgh, const float* __restrict__ bgh,
                                 const float* __restrict__ Wgc, const float* __restrict__ bgc,
                                 __bf16* __restrict__ Gt) {
    int gid = blockIdx.x * blockDim.x + threadIdx.x;   // 2048*128 threads
    int m = gid & (NN - 1);
    int n = gid >> 11;                                  // 0..127
    const float* prev = (n < HH) ? h_cur : c_cur;
    const float* W    = (n < HH) ? Wgh   : Wgc;
    int np = n & (HH - 1);
    float s = (n < HH) ? bgh[np] : bgc[np];
    const float* pm = prev + m * HH;
#pragma unroll 8
    for (int h = 0; h < HH; ++h) s += pm[h] * W[h * HH + np];
    Gt[(size_t)n * NN + m] = (__bf16)s;
}

__global__ void zero_acc_kernel(float* __restrict__ acc, int n) {
    int gid = blockIdx.x * blockDim.x + threadIdx.x;
    if (gid < n) acc[gid] = 0.0f;
}

// ---------------------------------------------------------------------------
// The big GEMM: acc[2048,128] += A[2048,2048] @ G[2048,128]  (bf16 WMMA, f32 acc)
//   grid = (32 M-tiles of 64 rows) x (KSPLIT K-slices of 512)
//   block = 256 threads = 8 waves = 4 M-strips x 2 N-halves
//   each wave: 16 rows x 64 cols = 4 x v_wmma_f32_16x16x32_bf16 per K-step
//
// Double-buffered LDS B tile (128 cols x 32 k, padded stride 40 = 80B):
//   - next tile's global_load_b128s issue BEFORE the WMMA chain (latency
//     hidden under matrix math)
//   - all 4 B fragments ds_load'ed before the 4 back-to-back WMMAs
//   - one barrier per K-step (publish nxt + retire readers of cur)
// K-partials combined with global_atomic_add_f32.
// ---------------------------------------------------------------------------
__global__ void __launch_bounds__(256)
gemm_graph_kernel(const __bf16* __restrict__ Abf, const __bf16* __restrict__ Gt,
                  float* __restrict__ acc) {
    __shared__ __attribute__((aligned(16))) __bf16 ldsB[2][128 * 40];

    const int tid  = threadIdx.x;
    const int lane = tid & 31;
    const int wave = tid >> 5;
    const int mstrip = wave & 3;   // 0..3 -> 16-row strip
    const int nhalf  = wave >> 2;  // 0..1 -> 64-col half

    const int wg_m    = blockIdx.x * 64;
    const int k_begin = blockIdx.y * (NN / KSPLIT);
    const int NSTEPS  = (NN / KSPLIT) / 32;   // 16

    const int l15  = lane & 15;
    const int koff = (lane < 16) ? 0 : 8;     // 16-bit WMMA K-half per lane group
    const int rowA = wg_m + mstrip * 16 + l15;
    const __bf16* arow = Abf + (size_t)rowA * NN + k_begin + koff;

    // fixed per-thread cooperative-load coordinates (2 x b128 per thread)
    const int idx0 = tid, idx1 = tid + 256;               // 0..511 over 128 cols x 4 chunks
    const int colL0 = idx0 >> 2, kkL0 = (idx0 & 3) * 8;
    const int colL1 = idx1 >> 2, kkL1 = (idx1 & 3) * 8;
    const __bf16* gsrc0 = Gt + (size_t)colL0 * NN + k_begin + kkL0;
    const __bf16* gsrc1 = Gt + (size_t)colL1 * NN + k_begin + kkL1;
    __bf16* lds0[2] = { &ldsB[0][colL0 * 40 + kkL0], &ldsB[1][colL0 * 40 + kkL0] };
    __bf16* lds1[2] = { &ldsB[0][colL1 * 40 + kkL1], &ldsB[1][colL1 * 40 + kkL1] };

    // B-fragment LDS base per ntile (col = nhalf*64 + nt*16 + l15)
    const int bcol = nhalf * 64 + l15;

    v8f accv[4] = {};

    // stage first tile into buffer 0
    *(v8bf*)lds0[0] = *(const v8bf*)(gsrc0);
    *(v8bf*)lds1[0] = *(const v8bf*)(gsrc1);
    __syncthreads();

    int cur = 0;
    for (int s = 0; s < NSTEPS; ++s) {
        const int krel = s * 32;

        // prefetch next B tile into registers (overlaps with WMMA chain below)
        v8bf nxt0, nxt1;
        const bool has_next = (s + 1 < NSTEPS);
        if (has_next) {
            nxt0 = *(const v8bf*)(gsrc0 + krel + 32);
            nxt1 = *(const v8bf*)(gsrc1 + krel + 32);
        }

        // A fragment: lane<16 holds K{0..7,16..23}, lane>=16 holds K{8..15,24..31}
        bfrag a;
        a.h[0] = *(const v8bf*)(arow + krel);
        a.h[1] = *(const v8bf*)(arow + krel + 16);

        // load ALL B fragments first, then issue 4 back-to-back WMMAs
        bfrag b[4];
#pragma unroll
        for (int nt = 0; nt < 4; ++nt) {
            const __bf16* bp = &ldsB[cur][(bcol + nt * 16) * 40 + koff];
            b[nt].h[0] = *(const v8bf*)(bp);
            b[nt].h[1] = *(const v8bf*)(bp + 16);
        }
#pragma unroll
        for (int nt = 0; nt < 4; ++nt) {
            accv[nt] = __builtin_amdgcn_wmma_f32_16x16x32_bf16(
                false, a.v, false, b[nt].v, (short)0, accv[nt], false, false);
        }

        if (has_next) {
            const int nb = cur ^ 1;
            *(v8bf*)lds0[nb] = nxt0;
            *(v8bf*)lds1[nb] = nxt1;
            __syncthreads();   // publish nxt; retire all readers of cur
            cur = nb;
        }
    }

    // epilogue: C/D layout -> VGPR r: M = r + (lane<16?0:8), N = lane&15
#pragma unroll
    for (int nt = 0; nt < 4; ++nt) {
        int col = nhalf * 64 + nt * 16 + l15;
#pragma unroll
        for (int r = 0; r < 8; ++r) {
            int row = wg_m + mstrip * 16 + r + ((lane < 16) ? 0 : 8);
            atomicAdd(&acc[(size_t)row * 128 + col], accv[nt][r]);
        }
    }
}

__global__ void tanh_acc_kernel(float* __restrict__ acc, int n) {
    int gid = blockIdx.x * blockDim.x + threadIdx.x;
    if (gid < n) acc[gid] = tanhf(acc[gid]);
}

// ---------------------------------------------------------------------------
// Keras LSTMCell (gate order i,f,g,o): one thread per (node m, unit j)
// ---------------------------------------------------------------------------
__global__ void lstm_gates_kernel(const float* __restrict__ inputs,
                                  const float* __restrict__ h_prev, const float* __restrict__ c_prev,
                                  const float* __restrict__ Wk, const float* __restrict__ Wr,
                                  const float* __restrict__ bias,
                                  float* __restrict__ h_lstm, float* __restrict__ c_lstm, int t) {
    int gid = blockIdx.x * blockDim.x + threadIdx.x;   // 2048*64
    int m = gid >> 6, j = gid & 63;
    const float* x = inputs + (size_t)m * TT * DD + (size_t)t * DD;
    float zi = bias[j], zf = bias[64 + j], zg = bias[128 + j], zo = bias[192 + j];
#pragma unroll
    for (int d = 0; d < DD; ++d) {
        float xv = x[d];
        const float* w = Wk + d * 256;
        zi += xv * w[j]; zf += xv * w[64 + j]; zg += xv * w[128 + j]; zo += xv * w[192 + j];
    }
    const float* hp = h_prev + m * HH;
#pragma unroll 8
    for (int h = 0; h < HH; ++h) {
        float hv = hp[h];
        const float* w = Wr + h * 256;
        zi += hv * w[j]; zf += hv * w[64 + j]; zg += hv * w[128 + j]; zo += hv * w[192 + j];
    }
    float cp = c_prev[m * HH + j];
    float cl = sigmoidf(zf) * cp + sigmoidf(zi) * tanhf(zg);
    float hl = sigmoidf(zo) * tanhf(cl);
    c_lstm[m * HH + j] = cl;
    h_lstm[m * HH + j] = hl;
}

// ---------------------------------------------------------------------------
// h_new = sigmoid(h_lstm@W_h_cur + h_graph@W_h_prev + b_h), same for c.
// acc already tanh'd: cols 0..63 = h_graph row, 64..127 = c_graph row.
// Writes outputs [N,T,H] and the carry buffers (safe in-place: all readers of
// the carry for this step already ran earlier in the stream).
// ---------------------------------------------------------------------------
__global__ void combine_kernel(const float* __restrict__ h_lstm, const float* __restrict__ c_lstm,
                               const float* __restrict__ acc,
                               const float* __restrict__ Whc, const float* __restrict__ Whp,
                               const float* __restrict__ bh,
                               const float* __restrict__ Wcc, const float* __restrict__ Wcp,
                               const float* __restrict__ bc,
                               float* __restrict__ out_h, float* __restrict__ out_c,
                               float* __restrict__ h_cur, float* __restrict__ c_cur, int t) {
    int gid = blockIdx.x * blockDim.x + threadIdx.x;   // 2048*64
    int m = gid >> 6, j = gid & 63;
    const float* hl = h_lstm + m * HH;
    const float* cl = c_lstm + m * HH;
    const float* am = acc + (size_t)m * 128;
    float sh = bh[j], sc = bc[j];
#pragma unroll 8
    for (int h = 0; h < HH; ++h) {
        sh += hl[h] * Whc[h * HH + j] + am[h]      * Whp[h * HH + j];
        sc += cl[h] * Wcc[h * HH + j] + am[64 + h] * Wcp[h * HH + j];
    }
    float hn = sigmoidf(sh);
    float cn = sigmoidf(sc);
    size_t o = (size_t)m * TT * HH + (size_t)t * HH + j;
    out_h[o] = hn;
    out_c[o] = cn;
    h_cur[m * HH + j] = hn;
    c_cur[m * HH + j] = cn;
}

// ---------------------------------------------------------------------------
extern "C" void kernel_launch(void* const* d_in, const int* in_sizes, int n_in,
                              void* d_out, int out_size, void* d_ws, size_t ws_size,
                              hipStream_t stream) {
    const float* inputs   = (const float*)d_in[0];
    const float* h_init   = (const float*)d_in[1];
    const float* c_init   = (const float*)d_in[2];
    const float* A        = (const float*)d_in[3];
    const float* Wgh      = (const float*)d_in[4];
    const float* bgh      = (const float*)d_in[5];
    const float* Wgc      = (const float*)d_in[6];
    const float* bgc      = (const float*)d_in[7];
    const float* Whc      = (const float*)d_in[8];
    const float* Whp      = (const float*)d_in[9];
    const float* bh       = (const float*)d_in[10];
    const float* Wcc      = (const float*)d_in[11];
    const float* Wcp      = (const float*)d_in[12];
    const float* bc       = (const float*)d_in[13];
    const float* lstm_k   = (const float*)d_in[14];
    const float* lstm_r   = (const float*)d_in[15];
    const float* lstm_b   = (const float*)d_in[16];

    float* out_h = (float*)d_out;
    float* out_c = out_h + (size_t)NN * TT * HH;

    // workspace carve-up (~11.5 MB)
    char* ws = (char*)d_ws;
    __bf16* Abf   = (__bf16*)ws;                          ws += (size_t)NN * NN * sizeof(__bf16); // 8 MB
    __bf16* Gt    = (__bf16*)ws;                          ws += (size_t)128 * NN * sizeof(__bf16);
    float*  acc   = (float*)ws;                           ws += (size_t)NN * 128 * sizeof(float);
    float*  h_cur = (float*)ws;                           ws += (size_t)NN * HH * sizeof(float);
    float*  c_cur = (float*)ws;                           ws += (size_t)NN * HH * sizeof(float);
    float*  h_l   = (float*)ws;                           ws += (size_t)NN * HH * sizeof(float);
    float*  c_l   = (float*)ws;                           ws += (size_t)NN * HH * sizeof(float);

    // one-time A -> bf16 (L2-resident working set for the whole recurrence)
    convertA_kernel<<<(NN * NN) / 256, 256, 0, stream>>>(A, Abf, NN * NN);

    // initialize carries
    hipMemcpyAsync(h_cur, h_init, (size_t)NN * HH * sizeof(float), hipMemcpyDeviceToDevice, stream);
    hipMemcpyAsync(c_cur, c_init, (size_t)NN * HH * sizeof(float), hipMemcpyDeviceToDevice, stream);

    for (int t = 0; t < TT; ++t) {
        zero_acc_kernel<<<(NN * 128) / 256, 256, 0, stream>>>(acc, NN * 128);
        graph_pre_kernel<<<(NN * 128) / 256, 256, 0, stream>>>(h_cur, c_cur, Wgh, bgh, Wgc, bgc, Gt);
        gemm_graph_kernel<<<dim3(NN / 64, KSPLIT), 256, 0, stream>>>(Abf, Gt, acc);
        lstm_gates_kernel<<<(NN * HH) / 256, 256, 0, stream>>>(inputs, h_cur, c_cur,
                                                               lstm_k, lstm_r, lstm_b, h_l, c_l, t);
        tanh_acc_kernel<<<(NN * 128) / 256, 256, 0, stream>>>(acc, NN * 128);
        combine_kernel<<<(NN * HH) / 256, 256, 0, stream>>>(h_l, c_l, acc, Whc, Whp, bh,
                                                            Wcc, Wcp, bc, out_h, out_c, h_cur, c_cur, t);
    }
}
